// FeatureSteeredConvolutionKerasLayer_34222299414787
// MI455X (gfx1250) — matrix operand order repaired and checked by reading, not compiled, not run on hardware
//
#include <hip/hip_runtime.h>

// FeaStNet feature-steered convolution for MI455X (gfx1250, wave32, WMMA).
//
// Pipeline (all on `stream`):
//   k_cvt_data : data f32 -> bf16                         [V,64]
//   k_build_wt : var_w [M,C,O] -> Wt bf16 [512,64], row n = o*8+m, col = c
//   k_xu       : xu = data @ var_u                        [V,8] f32
//   k_bias     : out[v,o] = var_b[o]                      (re-init every call)
//   k_gemm_z   : z[v, o*8+m] = (W_m x_v)[o]  via v_wmma_f32_16x16x32_bf16,
//                B staged in LDS with global_load_async_to_lds_b128
//   k_edge     : softmax(q) per edge + out[src] += w_e * sum_m q_m z[dst,o,m]

typedef __attribute__((ext_vector_type(16))) __bf16 v16bf;
typedef __attribute__((ext_vector_type(8)))  __bf16 v8bf;
typedef __attribute__((ext_vector_type(8)))  float  v8f;

#define C_IN    64
#define M_HEADS 8
#define O_OUT   64
#define NZ      512   // M_HEADS * O_OUT

// ---------------------------------------------------------------- conversions

__global__ void k_cvt_data(const float* __restrict__ src, __bf16* __restrict__ dst, int n) {
    int i = blockIdx.x * 256 + threadIdx.x;
    if (i < n) dst[i] = (__bf16)src[i];
}

// Wt[n*64 + c] = var_w[m, c, o] with n = o*8 + m  (so z rows are [o][m] packed)
__global__ void k_build_wt(const float* __restrict__ w, __bf16* __restrict__ wt) {
    int gid = blockIdx.x * 256 + threadIdx.x;       // 512*64 = 32768 total
    int n = gid >> 6, c = gid & 63;
    int o = n >> 3,  m = n & 7;
    wt[gid] = (__bf16)w[(m * C_IN + c) * O_OUT + o];
}

// ------------------------------------------------------------------- xu, bias

__global__ void k_xu(const float* __restrict__ data, const float* __restrict__ u,
                     float* __restrict__ xu, int V) {
    int gid = blockIdx.x * 256 + threadIdx.x;
    if (gid >= V * M_HEADS) return;
    int v = gid >> 3, m = gid & 7;
    const float* dr = data + (size_t)v * C_IN;
    float acc = 0.f;
#pragma unroll
    for (int c = 0; c < C_IN; ++c) acc = fmaf(dr[c], u[c * M_HEADS + m], acc);
    xu[gid] = acc;
}

__global__ void k_bias(const float* __restrict__ b, float* __restrict__ out, int n) {
    int i = blockIdx.x * 256 + threadIdx.x;
    if (i < n) out[i] = b[i & (O_OUT - 1)];
}

// -------------------------------------------------- z = data @ Wt^T  (WMMA)

// One block (4 waves, 128 threads) computes a 16-row stripe of z across all
// 32 n-tiles. Wt (64 KB) is staged into LDS once per block via async copy;
// each wave keeps both A k-fragments in registers and sweeps 8 n-tiles
// (16 WMMAs per wave, one live accumulator at a time).
__global__ void k_gemm_z(const __bf16* __restrict__ dbf, const __bf16* __restrict__ wt,
                         __bf16* __restrict__ z, int V) {
    __shared__ __align__(1024) __bf16 wsm[NZ * C_IN];   // 64 KB

    const int tid   = threadIdx.x;       // 0..127
    const int lane  = tid & 31;
    const int wave  = tid >> 5;          // 0..3
    const int r     = lane & 15;         // row-of-tile for A, col-of-tile for B/C
    const int khalf = lane >> 4;         // which K half-group this lane holds
    const int mtile = blockIdx.x;

    // ---- stage Wt into LDS with the CDNA5 async copy engine (ASYNCcnt) ----
    {
        const unsigned lds_base = (unsigned)(uintptr_t)&wsm[0];
#pragma unroll
        for (int it = 0; it < 32; ++it) {               // 32 * 128 * 16B = 64 KB
            unsigned boff = (unsigned)(it * 2048 + tid * 16);
            const char* g = (const char*)wt + boff;
            asm volatile("global_load_async_to_lds_b128 %0, %1, off"
                         :: "v"(lds_base + boff), "v"(g) : "memory");
        }
        asm volatile("s_wait_asynccnt 0x0" ::: "memory");
        __syncthreads();
    }

    int row = mtile * 16 + r;
    const bool full = (mtile * 16 + 16) <= V;           // whole-tile guard (uniform)
    if (row >= V) row = V - 1;                          // clamp loads only
    const __bf16* __restrict__ arow = dbf + (size_t)row * C_IN;

    // A fragments for kk = 0 and kk = 32 (per lane one contiguous
    // 16-element run at K kk+khalf*16, matching B)
    v16bf a[2];
#pragma unroll
    for (int s = 0; s < 2; ++s) {
        a[s] = *(const v16bf*)(arow + s * 32 + khalf * 16);
    }

#pragma unroll
    for (int nt = 0; nt < 8; ++nt) {
        const int ntile = wave * 8 + nt;                // 0..31
        const __bf16* brow = &wsm[(ntile * 16 + r) * C_IN];

        v8f acc = {};
#pragma unroll
        for (int s = 0; s < 2; ++s) {
            // B 32x16: lane = column n, elems 0..15 = K kk + khalf*16 + {0..15}
            v16bf b = *(const v16bf*)(brow + s * 32 + khalf * 16);
            acc = __builtin_amdgcn_wmma_f32_16x16x32_bf16(
                      false, a[s], false, b, (short)0, acc, false, false);
        }

        // C/D layout: VGPR i -> M = i + 8*khalf, N = r
        const size_t col = (size_t)(ntile * 16 + r);
        if (full) {
#pragma unroll
            for (int i = 0; i < 8; ++i)
                z[(size_t)(mtile * 16 + khalf * 8 + i) * NZ + col] = (__bf16)acc[i];
        } else {
            for (int i = 0; i < 8; ++i) {
                int orow = mtile * 16 + khalf * 8 + i;
                if (orow < V) z[(size_t)orow * NZ + col] = (__bf16)acc[i];
            }
        }
    }
}

// --------------------------------------------------------------- edge kernel

// 256 threads = 4 edges x 64 lanes (one lane per output channel o).
__global__ void k_edge(const int* __restrict__ esrc, const int* __restrict__ edst,
                       const float* __restrict__ ew, const float* __restrict__ xu,
                       const float* __restrict__ vc, const __bf16* __restrict__ z,
                       float* __restrict__ out, int E) {
    __shared__ float q_sh[4][M_HEADS];
    const int slot = threadIdx.x >> 6;
    const int t    = threadIdx.x & 63;
    const long long e = (long long)blockIdx.x * 4 + slot;
    const bool valid = (e < E);

    int src = 0, dst = 0; float w = 0.f;
    if (valid) { src = esrc[e]; dst = edst[e]; w = ew[e]; }

    if (valid && t == 0) {   // one lane computes the 8-way softmax per edge
        float lg[M_HEADS];
        float mx = -3.0e38f;
#pragma unroll
        for (int m = 0; m < M_HEADS; ++m) {
            lg[m] = xu[(size_t)src * M_HEADS + m] - xu[(size_t)dst * M_HEADS + m] + vc[m];
            mx = fmaxf(mx, lg[m]);
        }
        float s = 0.f;
#pragma unroll
        for (int m = 0; m < M_HEADS; ++m) { lg[m] = __expf(lg[m] - mx); s += lg[m]; }
        float inv = 1.0f / s;
#pragma unroll
        for (int m = 0; m < M_HEADS; ++m) q_sh[slot][m] = lg[m] * inv;
    }
    __syncthreads();
    if (!valid) return;

    // z row for dst: 1 KB, 64 lanes x 16 B fully coalesced
    v8bf zv = *(const v8bf*)(z + (size_t)dst * NZ + t * M_HEADS);
    float acc = 0.f;
#pragma unroll
    for (int m = 0; m < M_HEADS; ++m) acc = fmaf(q_sh[slot][m], (float)zv[m], acc);

    atomicAdd(out + (size_t)src * O_OUT + t, w * acc);
}

// -------------------------------------------------------------------- launch

extern "C" void kernel_launch(void* const* d_in, const int* in_sizes, int n_in,
                              void* d_out, int out_size, void* d_ws, size_t ws_size,
                              hipStream_t stream) {
    const float* data = (const float*)d_in[0];   // [V, 64]
    const int*   esrc = (const int*)  d_in[1];   // [E]
    const int*   edst = (const int*)  d_in[2];   // [E]
    const float* ew   = (const float*)d_in[3];   // [E]
    const float* vu   = (const float*)d_in[4];   // [64, 8]
    const float* vc   = (const float*)d_in[5];   // [8]
    const float* vw   = (const float*)d_in[6];   // [8, 64, 64]
    const float* vb   = (const float*)d_in[7];   // [64]
    float* out = (float*)d_out;                  // [V, 64]

    const int V = in_sizes[0] / C_IN;
    const int E = in_sizes[1];

    // Workspace carve-up (256B aligned): xu | data_bf | Wt | z_bf  (~119 MB)
    char* ws = (char*)d_ws;
    size_t off = 0;
    auto carve = [&](size_t bytes) {
        void* p = ws + off;
        off = (off + bytes + 255) & ~(size_t)255;
        return p;
    };
    float*  xu  = (float*) carve((size_t)V * M_HEADS * sizeof(float));
    __bf16* dbf = (__bf16*)carve((size_t)V * C_IN * sizeof(__bf16));
    __bf16* wt  = (__bf16*)carve((size_t)NZ * C_IN * sizeof(__bf16));
    __bf16* z   = (__bf16*)carve((size_t)V * NZ * sizeof(__bf16));
    (void)ws_size; (void)n_in; (void)out_size;

    k_cvt_data<<<(V * C_IN + 255) / 256, 256, 0, stream>>>(data, dbf, V * C_IN);
    k_build_wt<<<(NZ * C_IN) / 256, 256, 0, stream>>>(vw, wt);
    k_xu<<<(V * M_HEADS + 255) / 256, 256, 0, stream>>>(data, vu, xu, V);
    k_bias<<<(V * O_OUT + 255) / 256, 256, 0, stream>>>(vb, out, V * O_OUT);

    k_gemm_z<<<(V + 15) / 16, 128, 0, stream>>>(dbf, wt, z, V);

    k_edge<<<(E + 3) / 4, 256, 0, stream>>>(esrc, edst, ew, xu, vc, z, out, E);
}
